// SLSTMEncoder_21294447854116
// MI455X (gfx1250) — matrix operand order repaired and checked by reading, not compiled, not run on hardware
//
#include <hip/hip_runtime.h>
#include <hip/hip_bf16.h>

// ---------------------------------------------------------------------------
// S-LSTM encoder for MI455X (gfx1250, wave32, WMMA).
// B=8, S=1024, H=256, L=7.  GEMMs in bf16 WMMA (f32 accum); pointwise in f32.
// Wave tile 64x64 (16 v_wmma per 32-wide K step, 4x fragment reuse).
// All GEMM shapes compile-time; index-based global addressing (no FLAT path).
// ---------------------------------------------------------------------------

typedef __bf16 bf16_t;
typedef __bf16 v16bf  __attribute__((ext_vector_type(16)));
typedef __bf16 bf16x8 __attribute__((ext_vector_type(8)));
typedef float  v8f    __attribute__((ext_vector_type(8)));

#define Bd 8
#define Sd 1024
#define Hd 256
#define BS 8192          // B*S tokens
#define N7 1792          // 7*H
#define KZ 768           // 3*H (h_l | h_r | h)
#define LAYERS 7

// ---------------------------------------------------------------------------
// A-fragment: 16x32 bf16 tile of one 16-row strip (contiguous K order).
// lanes 0-15 : K = kb+{0..15}                   (row = strip + lane)
// lanes 16-31: K = kb+{16..31}                  (row = strip + lane-16)
// base already includes half*16; loads at +kb and +kb+8.
// ---------------------------------------------------------------------------
__device__ __forceinline__ v16bf load_a_frag(const bf16_t* __restrict__ base,
                                             int kbase) {
  const bf16x8 lo = *reinterpret_cast<const bf16x8*>(base + kbase);
  const bf16x8 hi = *reinterpret_cast<const bf16x8*>(base + kbase + 8);
  v16bf a;
#pragma unroll
  for (int i = 0; i < 8; ++i) { a[i] = lo[i]; a[8 + i] = hi[i]; }
  return a;
}

// ---------------------------------------------------------------------------
// WMMA GEMM: D[M,NT] = A[M,KT] * W[KT,NT] (+epilogue adds).
// Wt is N-major: Wt[n*KT + k] -> B-fragment is one contiguous 32B load/lane.
// MODE 0: plain A (row stride KT).
// MODE 1: A rows are virtual [h_l | h_r | h] from hidden_bf (stride Hd);
//         K handled as 3 straight-line 256-wide regions; edge rows read from
//         a zeroed scratch page (no divergent branches around loads).
// ADD  0: none; 1: per-n bias (addRow[n]); 2: addFull[m*NT+n] + addRow[b*NT+n].
// Wave tile: 64(M) x 64(N); block = 4 waves stacked along N (64 x 256 tile).
// ---------------------------------------------------------------------------
template <int MODE, int ADD, int NT, int KT>
__global__ __launch_bounds__(128)
void wmma_gemm(const bf16_t* __restrict__ A,
               const bf16_t* __restrict__ Wt,
               const float*  __restrict__ addFull,
               const float*  __restrict__ addRow,
               const bf16_t* __restrict__ zpage,   // >=512B zeroed (MODE 1)
               float* __restrict__ D) {
  const int lane = threadIdx.x & 31;
  const int wave = threadIdx.x >> 5;
  const int half = lane >> 4;
  const int r    = lane & 15;
  const int m0   = blockIdx.y * 64;
  const int n0   = blockIdx.x * 256 + wave * 64;

  v8f acc[4][4];
#pragma unroll
  for (int im = 0; im < 4; ++im)
#pragma unroll
    for (int j = 0; j < 4; ++j)
      acc[im][j] = v8f{0.f,0.f,0.f,0.f,0.f,0.f,0.f,0.f};

  // B column bases (index off Wt; loads use immediate kb offsets)
  const bf16_t* colb[4];
#pragma unroll
  for (int j = 0; j < 4; ++j)
    colb[j] = Wt + (size_t)(n0 + j * 16 + r) * KT + half * 16;

  if (MODE == 0) {
    const bf16_t* rowb[4];
#pragma unroll
    for (int im = 0; im < 4; ++im)
      rowb[im] = A + (size_t)(m0 + im * 16 + r) * KT + half * 16;
#pragma unroll 2
    for (int kb = 0; kb < KT; kb += 32) {
      v16bf a[4];
#pragma unroll
      for (int im = 0; im < 4; ++im) a[im] = load_a_frag(rowb[im], kb);
#pragma unroll
      for (int j = 0; j < 4; ++j) {
        const v16bf b = *reinterpret_cast<const v16bf*>(colb[j] + kb);
#pragma unroll
        for (int im = 0; im < 4; ++im)
          acc[im][j] = __builtin_amdgcn_wmma_f32_16x16x32_bf16(
              false, a[im], false, b, (short)0, acc[im][j], false, false);
      }
    }
  } else {
    // regions: 0 -> h_l (row-1, zero page at s==0), 1 -> h_r (row+1, zero page
    // at s==Sd-1), 2 -> h (row).
#pragma unroll
    for (int region = 0; region < 3; ++region) {
      const bf16_t* rowb[4];
#pragma unroll
      for (int im = 0; im < 4; ++im) {
        const int row = m0 + im * 16 + r;
        const int s   = row & (Sd - 1);
        int src = row;
        bool z = false;
        if (region == 0)      { z = (s == 0);      src = row - 1; }
        else if (region == 1) { z = (s == Sd - 1); src = row + 1; }
        rowb[im] = (z ? zpage : A + (size_t)src * Hd) + half * 16;
      }
#pragma unroll 2
      for (int kb = 0; kb < Hd; kb += 32) {
        v16bf a[4];
#pragma unroll
        for (int im = 0; im < 4; ++im) a[im] = load_a_frag(rowb[im], kb);
#pragma unroll
        for (int j = 0; j < 4; ++j) {
          const v16bf b =
              *reinterpret_cast<const v16bf*>(colb[j] + region * Hd + kb);
#pragma unroll
          for (int im = 0; im < 4; ++im)
            acc[im][j] = __builtin_amdgcn_wmma_f32_16x16x32_bf16(
                false, a[im], false, b, (short)0, acc[im][j], false, false);
        }
      }
    }
  }

  // Epilogue: element (M = m0+im*16+half*8+i, N = n0+j*16+r) in acc[im][j][i]
#pragma unroll
  for (int im = 0; im < 4; ++im) {
#pragma unroll
    for (int i = 0; i < 8; ++i) {
      const int    mm   = m0 + im * 16 + half * 8 + i;
      const size_t rowD = (size_t)mm * NT;
      const size_t rowB = (size_t)(mm >> 10) * NT;
#pragma unroll
      for (int j = 0; j < 4; ++j) {
        const int nn = n0 + j * 16 + r;
        float v = acc[im][j][i];
        if (ADD == 2) v += addFull[rowD + nn] + addRow[rowB + nn];
        if (ADD == 1) v += addRow[nn];
        D[rowD + nn] = v;
      }
    }
  }
}

// ---------------------------------------------------------------------------
// Helpers / pointwise kernels
// ---------------------------------------------------------------------------
__device__ __forceinline__ float block_sum256(float v, float* sm) {
  const int t = threadIdx.x;
  sm[t] = v; __syncthreads();
  for (int off = 128; off > 0; off >>= 1) {
    if (t < off) sm[t] += sm[t + off];
    __syncthreads();
  }
  const float r = sm[0]; __syncthreads();
  return r;
}

__global__ __launch_bounds__(256)
void convert_bf16_kernel(const float* __restrict__ in, bf16_t* __restrict__ out, int n) {
  const int i = blockIdx.x * 256 + threadIdx.x;
  if (i < n) out[i] = (bf16_t)in[i];
}

// Wt[n*K + k] = (bf16) W[k*N + n]
__global__ __launch_bounds__(256)
void transpose_convert_kernel(const float* __restrict__ W, bf16_t* __restrict__ Wt,
                              int K, int N) {
  const int idx = blockIdx.x * 256 + threadIdx.x;
  if (idx >= K * N) return;
  const int n = idx / K, k = idx - n * K;
  Wt[idx] = (bf16_t)W[(size_t)k * N + n];
}

// WzT[n*768 + k] : k<512 -> Wc[k][n], else Wh[k-512][n]
__global__ __launch_bounds__(256)
void build_wzT_kernel(const float* __restrict__ Wc, const float* __restrict__ Wh,
                      bf16_t* __restrict__ WzT) {
  const int idx = blockIdx.x * 256 + threadIdx.x;
  if (idx >= N7 * KZ) return;
  const int n = idx / KZ, k = idx - n * KZ;
  const float v = (k < 2 * Hd) ? Wc[(size_t)k * N7 + n]
                               : Wh[(size_t)(k - 2 * Hd) * N7 + n];
  WzT[idx] = (bf16_t)v;
}

// out[b,h] = masked mean over s
__global__ __launch_bounds__(256)
void masked_mean_kernel(const float* __restrict__ x, const float* __restrict__ mask,
                        float* __restrict__ out) {
  const int b = blockIdx.x, h = threadIdx.x;
  float acc = 0.f, msum = 0.f;
  for (int s = 0; s < Sd; ++s) {
    const float mk = mask[b * Sd + s];
    acc  += x[((size_t)b * Sd + s) * Hd + h] * mk;
    msum += mk;
  }
  out[b * Hd + h] = acc / msum;
}

// per-batch: pi = g_h@Gi+gb, pa = h_avg@Ga; ig/og = sigmoid(LN(...)); igE = exp(ig)
__global__ __launch_bounds__(256)
void global_gates_kernel(const float* __restrict__ g_h, const float* __restrict__ h_avg,
                         const float* __restrict__ Gi, const float* __restrict__ gb,
                         const float* __restrict__ Ga, const float* __restrict__ lns,
                         const float* __restrict__ lnb,
                         float* __restrict__ igE, float* __restrict__ og_out,
                         float* __restrict__ pimid) {
  __shared__ float sm[256];
  __shared__ float gh_s[256], ha_s[256];
  const int b = blockIdx.x, h = threadIdx.x;
  gh_s[h] = g_h[b * Hd + h];
  ha_s[h] = h_avg[b * Hd + h];
  __syncthreads();
  float p0 = gb[h], p1 = gb[Hd + h], p2 = gb[2 * Hd + h], a0 = 0.f, a1 = 0.f;
  for (int k = 0; k < Hd; ++k) {
    const float g = gh_s[k], ha = ha_s[k];
    p0 += g * Gi[(size_t)k * (3 * Hd) + h];
    p1 += g * Gi[(size_t)k * (3 * Hd) + Hd + h];
    p2 += g * Gi[(size_t)k * (3 * Hd) + 2 * Hd + h];
    a0 += ha * Ga[(size_t)k * (2 * Hd) + h];
    a1 += ha * Ga[(size_t)k * (2 * Hd) + Hd + h];
  }
  const float xi = p0 + a0, xo = p2 + a1;
  float s1 = block_sum256(xi, sm), s2 = block_sum256(xi * xi, sm);
  float mean = s1 / 256.f, var = (s2 - 256.f * mean * mean) / 255.f;
  const float yi = lns[0 * Hd + h] * (xi - mean) * rsqrtf(var + 1e-6f) + lnb[0 * Hd + h];
  s1 = block_sum256(xo, sm); s2 = block_sum256(xo * xo, sm);
  mean = s1 / 256.f; var = (s2 - 256.f * mean * mean) / 255.f;
  const float yo = lns[2 * Hd + h] * (xo - mean) * rsqrtf(var + 1e-6f) + lnb[2 * Hd + h];
  igE[b * Hd + h]    = expf(1.f / (1.f + expf(-yi)));
  og_out[b * Hd + h] = 1.f / (1.f + expf(-yo));
  pimid[b * Hd + h]  = p1;
}

// per-token: hgE = exp(sigmoid(LN(pimid + ph)) + mss)
__global__ __launch_bounds__(256)
void hg_kernel(const float* __restrict__ ph, const float* __restrict__ pimid,
               const float* __restrict__ mask, const float* __restrict__ lns,
               const float* __restrict__ lnb, float* __restrict__ hgE) {
  __shared__ float sm[256];
  const int m = blockIdx.x, h = threadIdx.x;
  const int b = m >> 10, s = m & (Sd - 1);
  const float x = pimid[b * Hd + h] + ph[(size_t)m * Hd + h];
  const float s1 = block_sum256(x, sm), s2 = block_sum256(x * x, sm);
  const float mean = s1 / 256.f, var = (s2 - 256.f * mean * mean) / 255.f;
  const float y  = lns[1 * Hd + h] * (x - mean) * rsqrtf(var + 1e-6f) + lnb[1 * Hd + h];
  const float hg = 1.f / (1.f + expf(-y));
  const float mss = mask[b * Sd + s] * 1e25f - 1e25f;
  hgE[(size_t)m * Hd + h] = expf(hg + mss);
}

// sequence-axis softmax mix: g_c, g_h update
__global__ __launch_bounds__(256)
void gstate_kernel(const float* __restrict__ hgE, const float* __restrict__ cell,
                   const float* __restrict__ igE, const float* __restrict__ og,
                   float* __restrict__ g_c, float* __restrict__ g_h) {
  const int b = blockIdx.x, h = threadIdx.x;
  const float ie = igE[b * Hd + h];
  float den = ie, num = g_c[b * Hd + h] * ie;
  for (int s = 0; s < Sd; ++s) {
    const size_t idx = ((size_t)b * Sd + s) * Hd + h;
    const float e = hgE[idx];
    den += e;
    num += e * cell[idx];
  }
  const float gc = num / den;
  g_c[b * Hd + h] = gc;
  g_h[b * Hd + h] = og[b * Hd + h] * tanhf(gc);
}

// ghWg[b,n] = dot(g_h[b,:], Wg[:,n])
__global__ __launch_bounds__(256)
void ghwg_kernel(const float* __restrict__ g_h, const float* __restrict__ Wg,
                 float* __restrict__ ghWg) {
  __shared__ float gh_s[256];
  const int b = blockIdx.y;
  const int n = blockIdx.x * 256 + threadIdx.x;
  gh_s[threadIdx.x] = g_h[b * Hd + threadIdx.x];
  __syncthreads();
  float acc = 0.f;
  for (int k = 0; k < Hd; ++k) acc += gh_s[k] * Wg[(size_t)k * N7 + n];
  ghWg[(size_t)b * N7 + n] = acc;
}

// fused 7xLN + 6-gate softmax + cell/hidden update (cell double-buffered)
__global__ __launch_bounds__(256)
void pointwise_kernel(const float* __restrict__ z, const float* __restrict__ cellOld,
                      const float* __restrict__ g_c, const float* __restrict__ mask,
                      const float* __restrict__ lns, const float* __restrict__ lnb,
                      float* __restrict__ cellNew, float* __restrict__ hidden,
                      bf16_t* __restrict__ hidden_bf) {
  __shared__ float sm[256];
  const int m = blockIdx.x, h = threadIdx.x;
  const int b = m >> 10, s = m & (Sd - 1);
  float ln[7];
#pragma unroll
  for (int g = 0; g < 7; ++g) {
    const float x  = z[(size_t)m * N7 + g * Hd + h];
    const float s1 = block_sum256(x, sm), s2 = block_sum256(x * x, sm);
    const float mean = s1 / 256.f, var = (s2 - 256.f * mean * mean) / 255.f;
    ln[g] = lns[(3 + g) * Hd + h] * (x - mean) * rsqrtf(var + 1e-6f) + lnb[(3 + g) * Hd + h];
  }
  float e[6], esum = 0.f;
#pragma unroll
  for (int g = 0; g < 6; ++g) {
    const float sg = 1.f / (1.f + expf(-ln[g]));
    e[g] = expf(sg);
    esum += e[g];
  }
  const float inv = 1.f / esum;
  const float memv = tanhf(ln[6]);
  const size_t mh = (size_t)m * Hd + h;
  const float c_here = cellOld[mh];
  const float c_l = (s > 0)      ? cellOld[mh - Hd] : 0.f;
  const float c_r = (s < Sd - 1) ? cellOld[mh + Hd] : 0.f;
  // gate order: i, l, r, f, gl, o
  float cn = e[1] * inv * c_l + e[2] * inv * c_r + e[3] * inv * c_here
           + e[0] * inv * memv + e[4] * inv * g_c[b * Hd + h];
  const float o  = e[5] * inv;
  const float mk = mask[b * Sd + s];
  const float hd = o * tanhf(cn) * mk;
  cn *= mk;
  cellNew[mh]  = cn;
  hidden[mh]   = hd;
  hidden_bf[mh] = (bf16_t)hd;
}

// ---------------------------------------------------------------------------
// Host-side orchestration
// ---------------------------------------------------------------------------
extern "C" void kernel_launch(void* const* d_in, const int* in_sizes, int n_in,
                              void* d_out, int out_size, void* d_ws, size_t ws_size,
                              hipStream_t stream) {
  const float* inputs      = (const float*)d_in[0];
  const float* mask        = (const float*)d_in[1];
  const float* init_hidden = (const float*)d_in[2];
  const float* init_cell   = (const float*)d_in[3];
  const float* Wc  = (const float*)d_in[4];
  const float* Wh  = (const float*)d_in[5];
  const float* Wi  = (const float*)d_in[6];
  const float* bi  = (const float*)d_in[7];
  const float* Wg  = (const float*)d_in[8];
  const float* Gi  = (const float*)d_in[9];
  const float* gb  = (const float*)d_in[10];
  const float* Gh  = (const float*)d_in[11];
  const float* Ga  = (const float*)d_in[12];
  const float* lns = (const float*)d_in[13];
  const float* lnb = (const float*)d_in[14];

  char* wp = (char*)d_ws;
  auto alloc = [&](size_t bytes) -> char* {
    char* p = wp;
    wp += (bytes + 255) & ~(size_t)255;
    return p;
  };
  const size_t STATE_B  = (size_t)BS * Hd * sizeof(float);   // 8 MB
  float*  hidden    = (float*)alloc(STATE_B);
  float*  cellA     = (float*)alloc(STATE_B);
  float*  cellB     = (float*)alloc(STATE_B);
  bf16_t* hidden_bf = (bf16_t*)alloc((size_t)BS * Hd * sizeof(bf16_t));
  bf16_t* inputs_bf = (bf16_t*)alloc((size_t)BS * Hd * sizeof(bf16_t));
  bf16_t* WiT       = (bf16_t*)alloc((size_t)N7 * Hd * sizeof(bf16_t));
  bf16_t* WzT       = (bf16_t*)alloc((size_t)N7 * KZ * sizeof(bf16_t));
  bf16_t* GhT       = (bf16_t*)alloc((size_t)Hd * Hd * sizeof(bf16_t));
  bf16_t* zpage     = (bf16_t*)alloc((size_t)Hd * sizeof(bf16_t));      // 512 B zeros
  float*  proj      = (float*)alloc((size_t)BS * N7 * sizeof(float));   // 58.7 MB
  float*  zbuf      = (float*)alloc((size_t)BS * N7 * sizeof(float));   // 58.7 MB
  float*  ph        = (float*)alloc(STATE_B);
  float*  hgE       = (float*)alloc(STATE_B);
  float*  h_avg     = (float*)alloc((size_t)Bd * Hd * sizeof(float));
  float*  g_h       = (float*)alloc((size_t)Bd * Hd * sizeof(float));
  float*  g_c       = (float*)alloc((size_t)Bd * Hd * sizeof(float));
  float*  igE       = (float*)alloc((size_t)Bd * Hd * sizeof(float));
  float*  og        = (float*)alloc((size_t)Bd * Hd * sizeof(float));
  float*  pimid     = (float*)alloc((size_t)Bd * Hd * sizeof(float));
  float*  ghWg      = (float*)alloc((size_t)Bd * N7 * sizeof(float));

  // ---- one-time setup ----
  hipMemcpyAsync(hidden, init_hidden, STATE_B, hipMemcpyDeviceToDevice, stream);
  hipMemcpyAsync(cellA,  init_cell,   STATE_B, hipMemcpyDeviceToDevice, stream);
  hipMemsetAsync(zpage, 0, (size_t)Hd * sizeof(bf16_t), stream);

  {
    const int n = BS * Hd;
    convert_bf16_kernel<<<(n + 255) / 256, 256, 0, stream>>>(inputs, inputs_bf, n);
    convert_bf16_kernel<<<(n + 255) / 256, 256, 0, stream>>>(init_hidden, hidden_bf, n);
  }
  transpose_convert_kernel<<<(Hd * N7 + 255) / 256, 256, 0, stream>>>(Wi, WiT, Hd, N7);
  transpose_convert_kernel<<<(Hd * Hd + 255) / 256, 256, 0, stream>>>(Gh, GhT, Hd, Hd);
  build_wzT_kernel<<<(N7 * KZ + 255) / 256, 256, 0, stream>>>(Wc, Wh, WzT);

  masked_mean_kernel<<<Bd, 256, 0, stream>>>(init_hidden, mask, g_h);
  masked_mean_kernel<<<Bd, 256, 0, stream>>>(init_cell,   mask, g_c);

  // proj_in = inputs @ Wi + bi  (bias via addRow)
  {
    dim3 grid(N7 / 256, BS / 64);
    wmma_gemm<0, 1, N7, Hd><<<grid, 128, 0, stream>>>(
        inputs_bf, WiT, nullptr, bi, nullptr, proj);
  }

  // ---- 7 layers ----
  float* cellOld = cellA;
  float* cellNew = cellB;
  for (int layer = 0; layer < LAYERS; ++layer) {
    // sentence-state path
    masked_mean_kernel<<<Bd, 256, 0, stream>>>(hidden, mask, h_avg);
    global_gates_kernel<<<Bd, 256, 0, stream>>>(g_h, h_avg, Gi, gb, Ga, lns, lnb,
                                                igE, og, pimid);
    {
      dim3 grid(Hd / 256, BS / 64);   // ph = hidden @ Gh
      wmma_gemm<0, 0, Hd, Hd><<<grid, 128, 0, stream>>>(
          hidden_bf, GhT, nullptr, nullptr, nullptr, ph);
    }
    hg_kernel<<<BS, 256, 0, stream>>>(ph, pimid, mask, lns, lnb, hgE);
    gstate_kernel<<<Bd, 256, 0, stream>>>(hgE, cellOld, igE, og, g_c, g_h);
    {
      dim3 grid(N7 / 256, Bd);        // ghWg = g_h @ Wg
      ghwg_kernel<<<grid, 256, 0, stream>>>(g_h, Wg, ghWg);
    }
    // z = [h_l,h_r,h] @ [Wc;Wh] + proj_in + ghWg (broadcast)
    {
      dim3 grid(N7 / 256, BS / 64);
      wmma_gemm<1, 2, N7, KZ><<<grid, 128, 0, stream>>>(
          hidden_bf, WzT, proj, ghWg, zpage, zbuf);
    }
    // gates + state update
    pointwise_kernel<<<BS, 256, 0, stream>>>(zbuf, cellOld, g_c, mask, lns, lnb,
                                             cellNew, hidden, hidden_bf);
    // swap cell buffers
    float* t = cellOld; cellOld = cellNew; cellNew = t;
  }

  hipMemcpyAsync(d_out, hidden, STATE_B, hipMemcpyDeviceToDevice, stream);
}
